// UniversalRNN_15169824489766
// MI455X (gfx1250) — matrix operand order repaired and checked by reading, not compiled, not run on hardware
//
#include <hip/hip_runtime.h>
#include <hip/hip_bf16.h>

typedef __attribute__((ext_vector_type(16))) _Float16 v16h;
typedef __attribute__((ext_vector_type(8)))  float    v8f;
typedef __attribute__((ext_vector_type(8)))  int      v8i;

namespace {
constexpr int Bn = 2048;
constexpr int Tn = 512;
constexpr int Hn = 64;
constexpr int ROWS_PER_WAVE   = 16;
constexpr int WAVES_PER_BLOCK = 4;
}

__device__ __forceinline__ int pk_f16(float a, float b) {
  // v_cvt_pk_rtz_f16_f32 -> two halves packed in one dword
  return __builtin_bit_cast(int, __builtin_amdgcn_cvt_pkrtz(a, b));
}

__device__ __forceinline__ float tanh_fast(float x) {
#if __has_builtin(__builtin_amdgcn_tanhf)
  return __builtin_amdgcn_tanhf(x);          // v_tanh_f32 (trans op)
#elif __has_builtin(__builtin_amdgcn_tanh_f32)
  return __builtin_amdgcn_tanh_f32(x);
#else
  return tanhf(x);
#endif
}

__device__ __forceinline__ v8f wmma_step(v16h a, v16h b, v8f c) {
  return __builtin_amdgcn_wmma_f32_16x16x32_f16(
      /*neg_a=*/false, a, /*neg_b=*/false, b,
      /*c_mod=*/(short)0, c, /*reuse_a=*/false, /*reuse_b=*/false);
}

// Build one 16x32 f16 A-fragment of row-major f32 weight W[64][64].
// A layout: lane<16: M=lane, K = 32*kc + {0..15};
//           lane>=16: M=lane-16, K = 32*kc + {16..31}.
__device__ __forceinline__ v16h load_a_frag(const float* __restrict__ W,
                                            int mt, int kc, int lane) {
  const int m  = 16 * mt + (lane & 15);
  const int k0 = 32 * kc + ((lane >> 4) & 1) * 16;
  const float* row = W + m * Hn;
  v8i u;
#pragma unroll
  for (int j = 0; j < 4; ++j) {
    u[j]     = pk_f16(row[k0 + 2 * j],      row[k0 + 2 * j + 1]);
    u[j + 4] = pk_f16(row[k0 + 8 + 2 * j], row[k0 + 8 + 2 * j + 1]);
  }
  return __builtin_bit_cast(v16h, u);
}

// Convert two tanh'd f32 D-tiles (hidden rows [32kc,32kc+16) and
// [32kc+16,32kc+32), batch cols = lane%16) into one f16 B-fragment
// (K = 32 hidden values x N = 16 batch values).
// D layout: element(m,n): vgpr=m%8, lane = n + 16*(m/8).
// B layout: element(k,n): vgpr=(k%16)/2, half=k%2, lane = n + 16*(k/16).
__device__ __forceinline__ v16h make_b_frag(const v8f& tA, const v8f& tB, bool lo) {
  v8i u;
#pragma unroll
  for (int jj = 0; jj < 4; ++jj) {
    const int Pa = pk_f16(tA[2 * jj], tA[2 * jj + 1]);   // k%16 in [0,8)  of tile A
    const int Pb = pk_f16(tB[2 * jj], tB[2 * jj + 1]);   // k%16 in [0,8)  of tile B
    const int M  = lo ? Pb : Pa;
    const int S  = __shfl_xor(M, 16, 32);  // lanes<16 get Pa from lane+16; lanes>=16 get Pb from lane-16
    u[jj]     = lo ? Pa : S;               // k in [32kc,   32kc+8)  / [32kc+16, 32kc+24)
    u[jj + 4] = lo ? S  : Pb;              // k in [32kc+8, 32kc+16) / [32kc+24, 32kc+32)
  }
  return __builtin_bit_cast(v16h, u);
}

__global__ __launch_bounds__(32 * WAVES_PER_BLOCK, 1)
void rnn2_wmma_kernel(const float* __restrict__ x,
                      const float* __restrict__ Wih0, const float* __restrict__ Whh0,
                      const float* __restrict__ bih0, const float* __restrict__ bhh0,
                      const float* __restrict__ Wih1, const float* __restrict__ Whh1,
                      const float* __restrict__ bih1, const float* __restrict__ bhh1,
                      const float* __restrict__ Wfc,  const float* __restrict__ bfc,
                      float* __restrict__ out) {
  const int lane  = threadIdx.x & 31;
  const int wave  = threadIdx.x >> 5;
  const int tile  = blockIdx.x * WAVES_PER_BLOCK + wave;   // 0..127
  const int bbase = tile * ROWS_PER_WAVE;                  // batch base for this wave
  const bool lo   = lane < 16;
  const int hgrp  = (lane >> 4) & 1;

  // ---- one-time: weight A-fragments, register resident for all 512 steps ----
  v16h a_hh0[4][2], a_ih1[4][2], a_hh1[4][2];
#pragma unroll
  for (int mt = 0; mt < 4; ++mt)
#pragma unroll
    for (int kc = 0; kc < 2; ++kc) {
      a_hh0[mt][kc] = load_a_frag(Whh0, mt, kc, lane);
      a_ih1[mt][kc] = load_a_frag(Wih1, mt, kc, lane);
      a_hh1[mt][kc] = load_a_frag(Whh1, mt, kc, lane);
    }

  // ---- per-lane coefficient vectors in D-tile layout: h = 16*nt + i + 8*hgrp ----
  float w0v[4][8];
  v8f  b0f[4], b1f[4];
#pragma unroll
  for (int nt = 0; nt < 4; ++nt)
#pragma unroll
    for (int i = 0; i < 8; ++i) {
      const int h = 16 * nt + i + 8 * hgrp;
      w0v[nt][i] = Wih0[h];                 // IN == 1: rank-1 input projection
      b0f[nt][i] = bih0[h] + bhh0[h];
      b1f[nt][i] = bih1[h] + bhh1[h];
    }

  const float* xrow = x + (size_t)(bbase + (lane & 15)) * Tn;

  const v8i zi = {0, 0, 0, 0, 0, 0, 0, 0};
  const v16h zf16 = __builtin_bit_cast(v16h, zi);
  v8f zacc;
#pragma unroll
  for (int i = 0; i < 8; ++i) zacc[i] = 0.0f;

  v16h h0B[2], h1B[2];
  h1B[0] = h1B[1] = zf16;

  // ---- prologue: h0(t=0) = tanh(x_0*W_ih0 + b0)  (h0 initial state is 0) ----
  {
    const float x0 = xrow[0];
    v8f a0[4];
#pragma unroll
    for (int nt = 0; nt < 4; ++nt)
#pragma unroll
      for (int i = 0; i < 8; ++i) a0[nt][i] = tanh_fast(fmaf(x0, w0v[nt][i], b0f[nt][i]));
    h0B[0] = make_b_frag(a0[0], a0[1], lo);
    h0B[1] = make_b_frag(a0[2], a0[3], lo);
  }

  v8f acc1[4] = {zacc, zacc, zacc, zacc};    // tanh'd layer-1 state (D layout)

  // Software-pipelined main loop: iteration t computes layer-1(t) AND
  // layer-0(t+1); both depend only on h0(t) / h1(t-1), so all three WMMA
  // accumulator chains (depth 2 each) are mutually independent.
  float xc1 = xrow[1 & (Tn - 1)];            // x(t+1) for the in-loop layer-0
#pragma unroll 1
  for (int t = 0; t < Tn; ++t) {
    const float xn = xrow[(t + 2) & (Tn - 1)];   // prefetch 2 steps ahead

    v8f accA[4], accB[4], acc0[4];
#pragma unroll
    for (int nt = 0; nt < 4; ++nt) {
      // layer-1(t): split into two independent accumulators
      accA[nt] = wmma_step(a_ih1[nt][0], h0B[0], b1f[nt]);
      accA[nt] = wmma_step(a_ih1[nt][1], h0B[1], accA[nt]);
      accB[nt] = wmma_step(a_hh1[nt][0], h1B[0], zacc);
      accB[nt] = wmma_step(a_hh1[nt][1], h1B[1], accB[nt]);
      // layer-0(t+1): recurrent GEMM on h0(t)
      v8f xp;
#pragma unroll
      for (int i = 0; i < 8; ++i) xp[i] = fmaf(xc1, w0v[nt][i], b0f[nt][i]);
      acc0[nt] = wmma_step(a_hh0[nt][0], h0B[0], xp);
      acc0[nt] = wmma_step(a_hh0[nt][1], h0B[1], acc0[nt]);
    }

    // combine + activations
    v8f h0t[4];
#pragma unroll
    for (int nt = 0; nt < 4; ++nt)
#pragma unroll
      for (int i = 0; i < 8; ++i) {
        acc1[nt][i] = tanh_fast(accA[nt][i] + accB[nt][i]);   // h1(t)
        h0t[nt][i]  = tanh_fast(acc0[nt][i]);                 // h0(t+1)
      }

    h1B[0] = make_b_frag(acc1[0], acc1[1], lo);
    h1B[1] = make_b_frag(acc1[2], acc1[3], lo);
    h0B[0] = make_b_frag(h0t[0], h0t[1], lo);
    h0B[1] = make_b_frag(h0t[2], h0t[3], lo);

    xc1 = xn;
  }

  // ======== final FC: out[b] = sum_h h1_last[b][h] * W_fc[h] + b_fc ========
  float s = 0.0f;
#pragma unroll
  for (int nt = 0; nt < 4; ++nt)
#pragma unroll
    for (int i = 0; i < 8; ++i) {
      const int h = 16 * nt + i + 8 * hgrp;
      s = fmaf(acc1[nt][i], Wfc[h], s);
    }
  s += __shfl_xor(s, 16, 32);                 // combine the two hidden halves
  if (lo) out[bbase + lane] = s + bfc[0];
}

extern "C" void kernel_launch(void* const* d_in, const int* in_sizes, int n_in,
                              void* d_out, int out_size, void* d_ws, size_t ws_size,
                              hipStream_t stream) {
  (void)in_sizes; (void)n_in; (void)d_ws; (void)ws_size; (void)out_size;
  const float* x    = (const float*)d_in[0];
  const float* Wih0 = (const float*)d_in[1];
  const float* Whh0 = (const float*)d_in[2];
  const float* bih0 = (const float*)d_in[3];
  const float* bhh0 = (const float*)d_in[4];
  const float* Wih1 = (const float*)d_in[5];
  const float* Whh1 = (const float*)d_in[6];
  const float* bih1 = (const float*)d_in[7];
  const float* bhh1 = (const float*)d_in[8];
  const float* Wfc  = (const float*)d_in[9];
  const float* bfc  = (const float*)d_in[10];
  float* out = (float*)d_out;

  dim3 grid(Bn / (ROWS_PER_WAVE * WAVES_PER_BLOCK));   // 32 blocks
  dim3 block(32 * WAVES_PER_BLOCK);                    // 4 independent waves each
  hipLaunchKernelGGL(rnn2_wmma_kernel, grid, block, 0, stream,
                     x, Wih0, Whh0, bih0, bhh0, Wih1, Whh1, bih1, bhh1, Wfc, bfc, out);
}
